// DeformConv3d_8220567404793
// MI455X (gfx1250) — matrix operand order repaired and checked by reading, not compiled, not run on hardware
//
#include <hip/hip_runtime.h>
#include <math.h>

typedef __attribute__((ext_vector_type(16))) _Float16 v16h;
typedef __attribute__((ext_vector_type(8)))  float    v8f;
typedef __attribute__((ext_vector_type(4)))  unsigned u32x4;
typedef __attribute__((ext_vector_type(8)))  int      i32x8;
typedef __attribute__((ext_vector_type(4)))  int      i32x4;

#define DDIM 64
#define DHW  (DDIM * DDIM * DDIM)   // 262144
#define CIN  16
#define COUT 16
#define KTAP 27
#define KK   432                    // CIN * KTAP
#define KKP  448                    // padded to 14 * 32
#define NOFF 81                     // 3 * KTAP offset channels
#define NOFFP 96
#define NT1  6                      // N tiles in offset GEMM
#define KCH  14                     // K chunks of 32

// d_ws layout:
//   B1t f16 [NOFFP][KKP]  offset-conv weights, transposed   (86016 B)
//   B2t f16 [COUT][KKP]   final einsum weights, transposed  (14336 B)
//   xT  f32 [DHW][CIN]    position-major transpose of x     (16 MB, 128B-aligned)
#define B1T_ELEMS (NOFFP * KKP)
#define B2T_OFF   B1T_ELEMS
#define PREP_TOT  (B1T_ELEMS + COUT * KKP)
#define XT_BYTE_OFF ((size_t)(PREP_TOT * 2))   // 100352, 128-aligned

// ---------------------------------------------------------------------------
// Prep 1: f32 weights -> f16, transposed + K-padded.
//   B1t[o][kk2], kk2 = kidx*16 + c : pkernel[kidx][c][o] = pkern[kk2*81 + o]
//   B2t[o][kk],  kk  = k*16 + c    : kernel[k][c][o]     = kern[kk*16 + o]
// ---------------------------------------------------------------------------
__global__ void prep_weights(const float* __restrict__ kern,
                             const float* __restrict__ pkern,
                             _Float16* __restrict__ ws) {
  int i = blockIdx.x * blockDim.x + threadIdx.x;
  if (i >= PREP_TOT) return;
  float v = 0.f;
  if (i < B1T_ELEMS) {
    int o = i / KKP, kk = i - o * KKP;
    if (o < NOFF && kk < KK) v = pkern[kk * NOFF + o];
  } else {
    int j = i - B1T_ELEMS;
    int o = j / KKP, kk = j - o * KKP;
    if (kk < KK) v = kern[kk * COUT + o];
  }
  ws[i] = (_Float16)v;
}

// ---------------------------------------------------------------------------
// Prep 2: x [16][64^3] -> xT [64^3][16] (channel-contiguous gather target)
// ---------------------------------------------------------------------------
__global__ void transpose_x(const float* __restrict__ x, float* __restrict__ xT) {
  int pos = blockIdx.x * blockDim.x + threadIdx.x;
  if (pos >= DHW) return;
  float4* q = (float4*)(xT + (size_t)pos * CIN);
#pragma unroll
  for (int g = 0; g < 4; ++g) {
    float4 v;
    v.x = x[(g * 4 + 0) * DHW + pos];
    v.y = x[(g * 4 + 1) * DHW + pos];
    v.z = x[(g * 4 + 2) * DHW + pos];
    v.w = x[(g * 4 + 3) * DHW + pos];
    q[g] = v;
  }
}

// A-fragment for v_wmma_f32_16x16x32_f16 (this code's contiguous k order, same as B loads):
//   lane L: M = L&15 ; K(j) = j + (L>>4)*16
__device__ __forceinline__ v16h frag_a(const _Float16* A, int t, int m, int hi) {
  v16h a;
#pragma unroll
  for (int j = 0; j < 16; ++j) {
    int kloc = j + hi * 16;
    a[j] = A[m * KKP + t * 32 + kloc];
  }
  return a;
}

__device__ __forceinline__ v16h cvt16(const float4* q) {
  float4 a0 = q[0], a1 = q[1], a2 = q[2], a3 = q[3];
  v16h h;
  h[0]=(_Float16)a0.x; h[1]=(_Float16)a0.y; h[2]=(_Float16)a0.z; h[3]=(_Float16)a0.w;
  h[4]=(_Float16)a1.x; h[5]=(_Float16)a1.y; h[6]=(_Float16)a1.z; h[7]=(_Float16)a1.w;
  h[8]=(_Float16)a2.x; h[9]=(_Float16)a2.y; h[10]=(_Float16)a2.z; h[11]=(_Float16)a2.w;
  h[12]=(_Float16)a3.x; h[13]=(_Float16)a3.y; h[14]=(_Float16)a3.z; h[15]=(_Float16)a3.w;
  return h;
}

// ---------------------------------------------------------------------------
// Fused deformable-conv3d: one wave per 16-position x-strip at fixed (z,y).
// ---------------------------------------------------------------------------
__global__ void __launch_bounds__(32)
deform_conv3d_wmma(const float* __restrict__ xT,
                   const _Float16* __restrict__ B1t,
                   const _Float16* __restrict__ B2t,
                   float* __restrict__ out) {
  __shared__ __align__(32) _Float16 Ash[16 * KKP];     // im2col / acc tile (14 KB)
  __shared__ __align__(32) _Float16 B2sh[COUT * KKP];  // TDM-staged weights (14 KB)
  __shared__ float offsh[16][NOFFP];                   // predicted offsets (6 KB)

  const int lane = threadIdx.x;           // single wave32
  const int ln16 = lane & 15;
  const int hi   = lane >> 4;
  const int x0   = blockIdx.x * 16;
  const int y    = blockIdx.y;
  const int z    = blockIdx.z;

  // ---- Kick off TDM: stage B2t (16x448 f16, row stride 448) into LDS.
  // D# per ISA 8.3/8.4: group0 = {count|flags, lds_addr, global_addr, type},
  // group1 = {data_size, tensor_dim0/1, tile_dim0/1/2, tensor_dim0_stride}.
  {
    unsigned ldsa = (unsigned)(uintptr_t)&B2sh[0];     // LDS aperture: low 32 bits
    unsigned long long ga = (unsigned long long)(uintptr_t)B2t;
    u32x4 g0;
    g0[0] = 1u;                                        // count=1, user descriptor
    g0[1] = ldsa;                                      // lds_addr (bytes)
    g0[2] = (unsigned)(ga & 0xffffffffu);              // global_addr[31:0]
    g0[3] = (unsigned)((ga >> 32) & 0x01ffffffu)       // global_addr[56:32]
          | (2u << 30);                                // type=2 ("image")
    i32x8 g1;
    g1[0] = (1 << 16);                                 // data_size=1 (2 bytes)
    g1[1] = (int)(448u << 16);                         // tensor_dim0=448 (lo16)
    g1[2] = (int)(16u << 16);                          // dim0 hi=0 | tensor_dim1=16
    g1[3] = (int)(448u << 16);                         // dim1 hi=0 | tile_dim0=448
    g1[4] = 16;                                        // tile_dim1=16, tile_dim2=0
    g1[5] = 448;                                       // tensor_dim0_stride lo32
    g1[6] = 0;                                         // stride hi | dim1_stride lo
    g1[7] = 0;
    i32x4 gz4 = {};
    i32x8 gz8 = {};
    __builtin_amdgcn_tensor_load_to_lds(g0, g1, gz4, gz4, gz8, 0);
  }

  // ---- Phase 1a: im2col tile -> LDS (f16), kk2 = kidx*16 + c (c contiguous)
  for (int p = lane; p < 16 * KTAP; p += 32) {
    int kidx = p >> 4;
    int m    = p & 15;
    int kz = kidx / 9, kr = kidx - kz * 9;
    int ky = kr / 3,   kx = kr - ky * 3;
    int zi = z + kz - 1, yi = y + ky - 1, xi = x0 + m + kx - 1;
    v16h h;
    if (zi >= 0 && zi < DDIM && yi >= 0 && yi < DDIM && xi >= 0 && xi < DDIM) {
      const float4* q = (const float4*)(xT + (size_t)((zi * DDIM + yi) * DDIM + xi) * CIN);
      h = cvt16(q);
    } else {
      v16h zv = {};
      h = zv;
    }
    *(v16h*)&Ash[m * KKP + (kidx << 4)] = h;
  }
  // zero K padding [432,448)
  for (int i = lane; i < 16 * (KKP - KK); i += 32)
    Ash[(i >> 4) * KKP + KK + (i & 15)] = (_Float16)0.f;
  __syncthreads();

  // ---- Phase 1b: offset GEMM (6 N-tiles x 14 K-chunks of WMMA)
  v8f acc1[NT1];
#pragma unroll
  for (int n = 0; n < NT1; ++n) { v8f zz = {}; acc1[n] = zz; }

  for (int t = 0; t < KCH; ++t) {
    v16h a = frag_a(Ash, t, ln16, hi);
#pragma unroll
    for (int n = 0; n < NT1; ++n) {
      const v16h b = *(const v16h*)(B1t + (n * 16 + ln16) * KKP + t * 32 + hi * 16);
      acc1[n] = __builtin_amdgcn_wmma_f32_16x16x32_f16(
          false, a, false, b, (short)0, acc1[n], false, false);
    }
  }
#pragma unroll
  for (int n = 0; n < NT1; ++n)
#pragma unroll
    for (int v = 0; v < 8; ++v)
      offsh[v + hi * 8][n * 16 + ln16] = acc1[n][v];
  __syncthreads();

  // ---- Phase 2: trilinear gather from xT -> acc tile (f16), kk = k*16 + c
  for (int km = lane; km < KTAP * 16; km += 32) {
    int k = km >> 4;
    int m = km & 15;
    int kz = k / 9, kr = k - kz * 9;
    int ky = kr / 3, kx = kr - ky * 3;
    float cz = (float)(z + kz - 1)      + offsh[m][k * 3 + 0];
    float cy = (float)(y + ky - 1)      + offsh[m][k * 3 + 1];
    float cx = (float)(x0 + m + kx - 1) + offsh[m][k * 3 + 2];
    float zf = floorf(cz), yf = floorf(cy), xf = floorf(cx);
    float fz = cz - zf, fy = cy - yf, fx = cx - xf;
    int z0 = (int)zf, y0 = (int)yf, xq = (int)xf;

    float accv[CIN];
#pragma unroll
    for (int c = 0; c < CIN; ++c) accv[c] = 0.f;

#pragma unroll
    for (int dz = 0; dz < 2; ++dz)
#pragma unroll
      for (int dy = 0; dy < 2; ++dy)
#pragma unroll
        for (int dx = 0; dx < 2; ++dx) {
          int zi = z0 + dz, yi = y0 + dy, xi = xq + dx;
          if (zi >= 0 && zi < DDIM && yi >= 0 && yi < DDIM &&
              xi >= 0 && xi < DDIM) {
            float w = (dz ? fz : 1.f - fz) * (dy ? fy : 1.f - fy) *
                      (dx ? fx : 1.f - fx);
            const float4* q =
                (const float4*)(xT + (size_t)((zi * DDIM + yi) * DDIM + xi) * CIN);
            float4 a0 = q[0], a1 = q[1], a2 = q[2], a3 = q[3];
            accv[0]  += w * a0.x; accv[1]  += w * a0.y;
            accv[2]  += w * a0.z; accv[3]  += w * a0.w;
            accv[4]  += w * a1.x; accv[5]  += w * a1.y;
            accv[6]  += w * a1.z; accv[7]  += w * a1.w;
            accv[8]  += w * a2.x; accv[9]  += w * a2.y;
            accv[10] += w * a2.z; accv[11] += w * a2.w;
            accv[12] += w * a3.x; accv[13] += w * a3.y;
            accv[14] += w * a3.z; accv[15] += w * a3.w;
          }
        }
    v16h h;
#pragma unroll
    for (int c = 0; c < CIN; ++c) h[c] = (_Float16)accv[c];
    *(v16h*)&Ash[m * KKP + (k << 4)] = h;
  }
  __syncthreads();

  // ---- Phase 3: output GEMM; B from TDM-staged LDS
  __builtin_amdgcn_s_wait_tensorcnt(0);
  v8f acc2 = {};
  for (int t = 0; t < KCH; ++t) {
    v16h a = frag_a(Ash, t, ln16, hi);
    const v16h b = *(const v16h*)&B2sh[ln16 * KKP + t * 32 + hi * 16];
    acc2 = __builtin_amdgcn_wmma_f32_16x16x32_f16(
        false, a, false, b, (short)0, acc2, false, false);
  }

  const int base = (z * DDIM + y) * DDIM + x0;   // out [16][64][64][64]
#pragma unroll
  for (int v = 0; v < 8; ++v)
    out[ln16 * DHW + base + v + hi * 8] = acc2[v];
}

extern "C" void kernel_launch(void* const* d_in, const int* in_sizes, int n_in,
                              void* d_out, int out_size, void* d_ws, size_t ws_size,
                              hipStream_t stream) {
  const float* x     = (const float*)d_in[0];   // [1,16,64,64,64]
  const float* kern  = (const float*)d_in[1];   // [27,16,16]
  const float* pkern = (const float*)d_in[2];   // [27,16,81]
  float* out = (float*)d_out;                   // [1,16,64,64,64]
  _Float16* wsB = (_Float16*)d_ws;
  float* xT = (float*)((char*)d_ws + XT_BYTE_OFF);

  prep_weights<<<(PREP_TOT + 255) / 256, 256, 0, stream>>>(kern, pkern, wsB);
  transpose_x<<<DHW / 256, 256, 0, stream>>>(x, xT);

  dim3 grid(DDIM / 16, DDIM, DDIM);             // (4, 64, 64)
  deform_conv3d_wmma<<<grid, 32, 0, stream>>>(xT, wsB, wsB + B2T_OFF, out);
}
